// motion_primitives_layer_61125974557562
// MI455X (gfx1250) — matrix operand order repaired and checked by reading, not compiled, not run on hardware
//
#include <hip/hip_runtime.h>

typedef __attribute__((ext_vector_type(16))) __bf16          v16bf;
typedef __attribute__((ext_vector_type(8)))  float           v8f;
typedef __attribute__((ext_vector_type(16))) unsigned short  v16u;
typedef __attribute__((ext_vector_type(8)))  unsigned short  v8u;

#define B_  16
#define C_  256
#define H_  128
#define W_  128
#define HW_ (H_ * W_)          // 16384
#define NBH (B_ * H_)          // 2048

// float -> bf16, round-to-nearest-even
__device__ __forceinline__ unsigned short f2bf_u(float f) {
  unsigned u = __builtin_bit_cast(unsigned, f);
  unsigned r = u + 0x7FFFu + ((u >> 16) & 1u);
  return (unsigned short)(r >> 16);
}

__device__ __forceinline__ v8f wmma_bf(v16u a, v16u b, v8f c) {
  return __builtin_amdgcn_wmma_f32_16x16x32_bf16(
      false, __builtin_bit_cast(v16bf, a),
      false, __builtin_bit_cast(v16bf, b),
      (short)0, c, false, false);
}

// ---------------------------------------------------------------------------
// Kernel 1: pooled prototypes. poolB[bh][slot(16)][c(256)] bf16.
// slots: 0..7 = P8 bins, 8..11 = P4 bins, 12..13 = P2 bins, 14..15 = zero.
// ---------------------------------------------------------------------------
__global__ __launch_bounds__(256) void pool_kernel(const float* __restrict__ x,
                                                   unsigned short* __restrict__ poolB) {
  int bh = blockIdx.x;
  int b = bh >> 7, h = bh & 127;
  int c = threadIdx.x;
  const float* row = x + ((size_t)(b * C_ + c) * H_ + h) * (size_t)W_;
  float b8[8];
#pragma unroll
  for (int p = 0; p < 8; p++) {
    const float4* r4 = (const float4*)(row + p * 16);
    float s = 0.f;
#pragma unroll
    for (int j = 0; j < 4; j++) { float4 v = r4[j]; s += (v.x + v.y) + (v.z + v.w); }
    b8[p] = s * 0.0625f;
  }
  unsigned short* dst = poolB + (size_t)bh * 4096 + c;
#pragma unroll
  for (int p = 0; p < 8; p++) dst[p * 256] = f2bf_u(b8[p]);
#pragma unroll
  for (int j = 0; j < 4; j++) dst[(8 + j) * 256] = f2bf_u(0.5f * (b8[2 * j] + b8[2 * j + 1]));
  dst[12 * 256] = f2bf_u(0.25f * (b8[0] + b8[1] + b8[2] + b8[3]));
  dst[13 * 256] = f2bf_u(0.25f * (b8[4] + b8[5] + b8[6] + b8[7]));
  dst[14 * 256] = 0;
  dst[15 * 256] = 0;
}

// ---------------------------------------------------------------------------
// Kernel 2: G_i = Wf_i^T @ Wp_i  (bf16), T_i = Wo_i @ Wp_i (f32 temp)
// ---------------------------------------------------------------------------
__global__ __launch_bounds__(256) void wprep1(const float* __restrict__ Wp,
                                              const float* __restrict__ Wf,
                                              const float* __restrict__ Wo,
                                              unsigned short* __restrict__ Gs,
                                              float* __restrict__ T) {
  int id = blockIdx.x * 256 + threadIdx.x;      // 6 * 65536 total
  int mat = id >> 16;
  int r = (id >> 8) & 255, cc = id & 255;
  if (mat < 3) {
    const float* wf = Wf + (size_t)mat * 65536;
    const float* wp = Wp + (size_t)mat * 65536;
    float s = 0.f;
    for (int o = 0; o < 256; o++) s += wf[o * 256 + r] * wp[o * 256 + cc];
    Gs[(size_t)mat * 65536 + r * 256 + cc] = f2bf_u(s);
  } else {
    int i = mat - 3;
    const float* wo = Wo + (size_t)i * 65536;
    const float* wp = Wp + (size_t)i * 65536;
    float s = 0.f;
    for (int m = 0; m < 256; m++) s += wo[r * 256 + m] * wp[m * 256 + cc];
    T[(size_t)i * 65536 + r * 256 + cc] = s;
  }
}

// K_i = W_fuse[:, i*256:(i+1)*256] @ T_i  (bf16)
__global__ __launch_bounds__(256) void wprep2(const float* __restrict__ Wfuse,
                                              const float* __restrict__ T,
                                              unsigned short* __restrict__ Ks) {
  int id = blockIdx.x * 256 + threadIdx.x;      // 3 * 65536 total
  int i = id >> 16;
  int r = (id >> 8) & 255, cc = id & 255;
  const float* t = T + (size_t)i * 65536;
  float s = 0.f;
  for (int m = 0; m < 256; m++) s += Wfuse[r * 768 + i * 256 + m] * t[m * 256 + cc];
  Ks[(size_t)i * 65536 + r * 256 + cc] = f2bf_u(s);
}

// ---------------------------------------------------------------------------
// Kernel 3: per-(b,h) rank factors via WMMA, 4 (b,h) rows per workgroup.
//   Aarr[bh][slot(16)][c(256)]  = G_br  @ pool   (bf16)
//   Bvarr[bh][o(256)][slot(16)] = K_br  @ pool   (bf16)
// Pool blocks staged to LDS via GLOBAL_LOAD_ASYNC_TO_LDS (ASYNCcnt path).
// Each weight A-fragment feeds 4 independent WMMA chains (latency hiding,
// 4x fewer weight fetches).
// ---------------------------------------------------------------------------
__global__ __launch_bounds__(256) void ab_kernel(const unsigned short* __restrict__ Gs,
                                                 const unsigned short* __restrict__ Ks,
                                                 const unsigned short* __restrict__ poolB,
                                                 unsigned short* __restrict__ Aarr,
                                                 unsigned short* __restrict__ Bvarr) {
  // rows [blk*16 + s] for blk 0..3; rows 64..79 stay zero (pad columns)
  __shared__ alignas(32) unsigned short lpool[80 * 256];
  int bh0 = blockIdx.x * 4;

  int lane = threadIdx.x & 31;
  int wv = threadIdx.x >> 5;

  // zero pad slots (ws is poisoned, these are read later)
  {
    int t = threadIdx.x;
#pragma unroll
    for (int blk = 0; blk < 4; blk++) {
      size_t bh = (size_t)(bh0 + blk);
      Aarr[bh * 4096 + 14 * 256 + t] = 0;
      Aarr[bh * 4096 + 15 * 256 + t] = 0;
      Bvarr[(bh * 256 + t) * 16 + 14] = 0;
      Bvarr[(bh * 256 + t) * 16 + 15] = 0;
    }
  }

  // async-copy 4 x 8KB pool blocks into LDS (waves 0..3, one block each):
  // 16 ops x 32 lanes x 16B = 8192B per block, tracked on ASYNCcnt.
  if (wv < 4) {
    unsigned lds0 = (unsigned)(uintptr_t)(&lpool[0]) + (unsigned)(wv * 8192) + lane * 16u;
    unsigned long long g0 =
        (unsigned long long)(uintptr_t)(poolB + (size_t)(bh0 + wv) * 4096) + lane * 16u;
#pragma unroll
    for (int it = 0; it < 16; ++it) {
      unsigned l = lds0 + (unsigned)(it * 512);
      unsigned long long g = g0 + (unsigned long long)(it * 512);
      asm volatile("global_load_async_to_lds_b128 %0, %1, off"
                   :: "v"(l), "v"(g) : "memory");
    }
    asm volatile("s_wait_asynccnt 0" ::: "memory");
  }
  // zero rows 64..79 (zero B-columns for pad slots)
  for (int t = threadIdx.x; t < 4096; t += 256) lpool[16384 + t] = 0;
  __syncthreads();

  int n = lane & 15, g = lane >> 4;
  bool isBv = (wv & 4) != 0;
  const unsigned short* Wmat = isBv ? Ks : Gs;

  const int OFF[3] = {12, 8, 0};
  const int PP[3]  = {2, 4, 8};

  for (int mt = (wv & 3); mt < 16; mt += 4) {
#pragma unroll
    for (int br = 0; br < 3; br++) {
      int P = PP[br], off = OFF[br];

      // preload all K-step A-fragments (16 b128 loads in flight)
      v16u afr[8];
#pragma unroll
      for (int kk = 0; kk < 8; kk++) {
        const unsigned short* ap = Wmat + (size_t)br * 65536 + (mt * 16 + n) * 256 + kk * 32 + 16 * g;
        v8u a0 = *(const v8u*)ap;
        v8u a1 = *(const v8u*)(ap + 8);
        afr[kk] = __builtin_shufflevector(a0, a1, 0,1,2,3,4,5,6,7,8,9,10,11,12,13,14,15);
      }

      // 4 independent accumulator chains share each A-fragment
      v8f acc0 = {}, acc1 = {}, acc2 = {}, acc3 = {};
#pragma unroll
      for (int kk = 0; kk < 8; kk++) {
        int rs0 = (n < P) ? (0 * 16 + off + n) : (64 + n);
        int rs1 = (n < P) ? (1 * 16 + off + n) : (64 + n);
        int rs2 = (n < P) ? (2 * 16 + off + n) : (64 + n);
        int rs3 = (n < P) ? (3 * 16 + off + n) : (64 + n);
        acc0 = wmma_bf(afr[kk], *(const v16u*)(lpool + rs0 * 256 + kk * 32 + 16 * g), acc0);
        acc1 = wmma_bf(afr[kk], *(const v16u*)(lpool + rs1 * 256 + kk * 32 + 16 * g), acc1);
        acc2 = wmma_bf(afr[kk], *(const v16u*)(lpool + rs2 * 256 + kk * 32 + 16 * g), acc2);
        acc3 = wmma_bf(afr[kk], *(const v16u*)(lpool + rs3 * 256 + kk * 32 + 16 * g), acc3);
      }

      if (n < P) {
        v8f accs[4] = {acc0, acc1, acc2, acc3};
#pragma unroll
        for (int blk = 0; blk < 4; blk++) {
          size_t bh = (size_t)(bh0 + blk);
          if (!isBv) {
            v8u ov;
#pragma unroll
            for (int r = 0; r < 8; r++) ov[r] = f2bf_u(accs[blk][r]);
            *(v8u*)(Aarr + (bh * 16 + off + n) * 256 + mt * 16 + 8 * g) = ov;
          } else {
#pragma unroll
            for (int r = 0; r < 8; r++)
              Bvarr[(bh * 256 + mt * 16 + 8 * g + r) * 16 + off + n] = f2bf_u(accs[blk][r]);
          }
        }
      }
    }
  }
}

// ---------------------------------------------------------------------------
// Kernel 4: fused per-row pass.  logits -> segmented softmax -> rank-14
// reconstruction -> residual.  One workgroup per (b,h); wave w = w-tile.
// ---------------------------------------------------------------------------
__global__ __launch_bounds__(256) void main_kernel(const float* __restrict__ x,
                                                   const unsigned short* __restrict__ Aarr,
                                                   const unsigned short* __restrict__ Bvarr,
                                                   float* __restrict__ out) {
  __shared__ alignas(32) unsigned short xT[128 * 264];   // [w][c], pad 8
  __shared__ alignas(32) unsigned short attL[8][256];    // per-wave [w(16)][p(16)]

  int bh = blockIdx.x;
  int b = bh >> 7, h = bh & 127;
  const float* xrow = x + ((size_t)b * C_ * H_ + h) * (size_t)W_;   // + c*HW + w

  // stage x row as bf16, transposed to [w][c]
  for (int t = threadIdx.x; t < 8192; t += 256) {
    int c = t >> 5;
    int w4 = (t & 31) << 2;
    float4 v = *(const float4*)(xrow + (size_t)c * HW_ + w4);
    int base = w4 * 264 + c;
    xT[base]       = f2bf_u(v.x);
    xT[base + 264] = f2bf_u(v.y);
    xT[base + 528] = f2bf_u(v.z);
    xT[base + 792] = f2bf_u(v.w);
  }
  __syncthreads();

  int lane = threadIdx.x & 31;
  int wt = threadIdx.x >> 5;          // w-tile 0..7
  int n = lane & 15, g = lane >> 4;

  // hoist B-fragments of the logit GEMM: Aarr[bh][p=n][c]
  const unsigned short* Abase = Aarr + ((size_t)bh * 16 + n) * 256 + 16 * g;
  v16u bfr[8];
#pragma unroll
  for (int kk = 0; kk < 8; kk++) bfr[kk] = *(const v16u*)(Abase + kk * 32);

  // logits: D[w, p] = sum_c x[c,w] * A[c,p]
  v8f acc = {};
#pragma unroll
  for (int kk = 0; kk < 8; kk++) {
    const unsigned short* ap = xT + (wt * 16 + n) * 264 + kk * 32 + 16 * g;
    v8u a0 = *(const v8u*)ap;
    v8u a1 = *(const v8u*)(ap + 8);
    v16u af = __builtin_shufflevector(a0, a1, 0,1,2,3,4,5,6,7,8,9,10,11,12,13,14,15);
    acc = wmma_bf(af, bfr[kk], acc);
  }

  // prefetch the 16 Bv fragments for the output GEMM (keeps loads in flight
  // through the softmax VALU work)
  v8u bvfr[16];
#pragma unroll
  for (int ot = 0; ot < 16; ot++)
    bvfr[ot] = *(const v8u*)(Bvarr + ((size_t)bh * 256 + ot * 16 + n) * 16 + 8 * g);

  // segmented softmax across lanes: p segments {0..7},{8..11},{12..13},{14..15}
  int seg = (n < 8) ? 8 : ((n < 12) ? 4 : 2);
  float att8[8];
#pragma unroll
  for (int r = 0; r < 8; r++) {
    float v = acc[r] * 0.0625f;                    // 1/sqrt(C)
    float mx = v;
    float t1 = __shfl_xor(mx, 1, 32); mx = (1 < seg) ? fmaxf(mx, t1) : mx;
    float t2 = __shfl_xor(mx, 2, 32); mx = (2 < seg) ? fmaxf(mx, t2) : mx;
    float t4 = __shfl_xor(mx, 4, 32); mx = (4 < seg) ? fmaxf(mx, t4) : mx;
    float e = __expf(v - mx);
    float s = e;
    float s1 = __shfl_xor(s, 1, 32); s = (1 < seg) ? s + s1 : s;
    float s2 = __shfl_xor(s, 2, 32); s = (2 < seg) ? s + s2 : s;
    float s4 = __shfl_xor(s, 4, 32); s = (4 < seg) ? s + s4 : s;
    att8[r] = e / s;
  }

  // 16x16 transpose via per-wave LDS region (same-wave DS ops are in-order)
#pragma unroll
  for (int r = 0; r < 8; r++) attL[wt][(r + 8 * g) * 16 + n] = f2bf_u(att8[r]);

  // out[o,w] = sum_p Bv[o,p] * att[w,p]   (K padded 14->32 with zeros)
  v16u batt = {};
  batt = __builtin_shufflevector(*(const v8u*)(&attL[wt][n * 16 + 8 * g]), (v8u){}, 0,1,2,3,4,5,6,7,8,9,10,11,12,13,14,15);
  v8f zc = {};
  v8u z8 = {};
  for (int ot = 0; ot < 16; ot++) {
    v16u af = __builtin_shufflevector(bvfr[ot], z8, 0,1,2,3,4,5,6,7,8,9,10,11,12,13,14,15);
    v8f d = wmma_bf(af, batt, zc);
    size_t base = (((size_t)b * 256 + ot * 16 + 8 * g) * 128 + h) * 128 + wt * 16 + n;
#pragma unroll
    for (int r = 0; r < 8; r++) {
      size_t idx = base + (size_t)r * HW_;
      out[idx] = d[r] + x[idx];
    }
  }
}

// ---------------------------------------------------------------------------
extern "C" void kernel_launch(void* const* d_in, const int* in_sizes, int n_in,
                              void* d_out, int out_size, void* d_ws, size_t ws_size,
                              hipStream_t stream) {
  const float* x     = (const float*)d_in[0];
  const float* Wp    = (const float*)d_in[1];
  const float* Wf    = (const float*)d_in[2];
  const float* Wo    = (const float*)d_in[3];
  const float* Wfuse = (const float*)d_in[4];
  float* out = (float*)d_out;

  char* ws = (char*)d_ws;
  unsigned short* Gs    = (unsigned short*)(ws + 0);                      // 384 KB
  unsigned short* Ks    = (unsigned short*)(ws + 393216);                 // 384 KB
  float*          T     = (float*)(ws + 786432);                          // 768 KB
  unsigned short* poolB = (unsigned short*)(ws + 1572864);                // 16 MB
  unsigned short* Aarr  = (unsigned short*)(ws + 1572864 + 16777216);     // 16 MB
  unsigned short* Bvarr = (unsigned short*)(ws + 1572864 + 2 * 16777216); // 16 MB

  pool_kernel<<<dim3(NBH), dim3(256), 0, stream>>>(x, poolB);
  wprep1<<<dim3(1536), dim3(256), 0, stream>>>(Wp, Wf, Wo, Gs, T);
  wprep2<<<dim3(768), dim3(256), 0, stream>>>(Wfuse, T, Ks);
  ab_kernel<<<dim3(NBH / 4), dim3(256), 0, stream>>>(Gs, Ks, poolB, Aarr, Bvarr);
  main_kernel<<<dim3(NBH), dim3(256), 0, stream>>>(x, Aarr, Bvarr, out);
}
